// OCNet_85220741087707
// MI455X (gfx1250) — matrix operand order repaired and checked by reading, not compiled, not run on hardware
//
#include <hip/hip_runtime.h>

typedef __attribute__((ext_vector_type(16))) _Float16 v16h;
typedef __attribute__((ext_vector_type(8)))  float    v8f;

#define NPTS   1600
#define HEADS  4
#define HD     64
#define VPITCH (NPTS + 16)   // padded V row: first 16 cols replicated at end

// Fragment union: v16h is two contiguous 16-byte runs per lane in all the
// layouts we use (A, B, and LDS P tiles), so fill via float4 / int loads.
union Frag16 {
  v16h  h;
  float4 f4[2];
  int    i[8];
};

// ---------------------------------------------------------------------------
// Projection: xf(1600x256) @ W(256x256) + b for q,k,v (fp32 math, f16 out).
// Q pre-scaled by 1/sqrt(O)=1/16. V stored transposed+padded: Vt[g][d][n],
// row length VPITCH with Vt[g][d][NPTS+i] = Vt[g][d][i] (wrap pad) so the
// rolled PV B-fragment loads are always contiguous.
// ---------------------------------------------------------------------------
__global__ __launch_bounds__(256) void proj_kernel(
    const float* __restrict__ x,
    const float* __restrict__ Wq, const float* __restrict__ bq,
    const float* __restrict__ Wk, const float* __restrict__ bk,
    const float* __restrict__ Wv, const float* __restrict__ bv,
    _Float16* __restrict__ Qh, _Float16* __restrict__ Kh,
    _Float16* __restrict__ Vt)
{
  __shared__ float xs[256];
  const int n = blockIdx.x;
  const int o = threadIdx.x;
  xs[o] = x[n * 256 + o];
  __syncthreads();
  float aq = bq[o], ak = bk[o], av = bv[o];
#pragma unroll 8
  for (int c = 0; c < 256; ++c) {
    const float xv = xs[c];
    aq = fmaf(xv, Wq[c * 256 + o], aq);
    ak = fmaf(xv, Wk[c * 256 + o], ak);
    av = fmaf(xv, Wv[c * 256 + o], av);
  }
  const int g = o >> 6, d = o & 63;
  Qh[(size_t)(g * NPTS + n) * HD + d] = (_Float16)(aq * 0.0625f);
  Kh[(size_t)(g * NPTS + n) * HD + d] = (_Float16)ak;
  Vt[(size_t)(g * HD + d) * VPITCH + n] = (_Float16)av;
  if (n < 16)  // wrap pad
    Vt[(size_t)(g * HD + d) * VPITCH + NPTS + n] = (_Float16)av;
}

// ---------------------------------------------------------------------------
// Attention: block = (head g, 16-query tile); 5 waves, wave w owns dw = w-2.
// Pass (dh,dw), head g contributes (after roll reindexing):
//   softmax_j( q[g][n].k[(g-dh)%4][j] / 16 ) * v[(g+dh)%4][(j+2*dw) mod N]
// dh=+2 and dh=-2 are identical (head roll mod 4) -> 4 passes, last weighted 2.
// Softmax computed without max-subtraction (scores bounded ~|5|: exact in f32),
// so no per-chunk cross-lane reductions are needed at all.
// ---------------------------------------------------------------------------
__global__ __launch_bounds__(160) void attn_kernel(
    const _Float16* __restrict__ Qh, const _Float16* __restrict__ Kh,
    const _Float16* __restrict__ Vt, float* __restrict__ out)
{
  __shared__ _Float16 Plds[5][16][32];   // per-wave probability tile (16x32)
  __shared__ float partial[5][16][64];   // per-wave output partials

  const int wid  = threadIdx.x >> 5;     // 0..4  (dw = wid-2)
  const int lane = threadIdx.x & 31;
  const int l16  = lane & 15;
  const int half = lane >> 4;
  const int g    = blockIdx.x & 3;
  const int qt   = blockIdx.x >> 2;
  const int qbase  = qt * 16;
  const int vshift = (2 * (wid - 2) + NPTS) % NPTS;  // even, in [0,NPTS)

  // --- resident Q A-fragments: per lane two 16B runs each ---
  Frag16 Qa0, Qa1;
  {
    const _Float16* qrow = Qh + (size_t)(g * NPTS + qbase + l16) * HD;
    Qa0.f4[0] = *(const float4*)(qrow + 16 * half);
    Qa0.f4[1] = *(const float4*)(qrow + 8 + 16 * half);
    Qa1.f4[0] = *(const float4*)(qrow + 32 + 16 * half);
    Qa1.f4[1] = *(const float4*)(qrow + 40 + 16 * half);
  }

  v8f accT[4];
#pragma unroll
  for (int t = 0; t < 4; ++t) accT[t] = v8f{};

  const int   dhs[4]  = {0, 1, -1, 2};
  const float wgts[4] = {1.0f, 1.0f, 1.0f, 2.0f};

  for (int pi = 0; pi < 4; ++pi) {
    const int dh = dhs[pi];
    const int gk = ((g - dh) % 4 + 4) % 4;   // key head
    const int gv = ((g + dh) % 4 + 4) % 4;   // value head

    float rsum[8];
    v8f acc[4];
#pragma unroll
    for (int r = 0; r < 8; ++r) rsum[r] = 0.0f;
#pragma unroll
    for (int t = 0; t < 4; ++t) acc[t] = v8f{};

    for (int chunk = 0; chunk < NPTS / 32; ++chunk) {
      const int kb = chunk * 32;

      // ---- scores: two 16-key tiles, 2 WMMAs each over hd=64 ----
      v8f s[2];
#pragma unroll
      for (int kt = 0; kt < 2; ++kt) {
        // B fragment = 16 contiguous halfs at row (key), col 16*half (B1: +32)
        const _Float16* ka =
            Kh + (size_t)(gk * NPTS + kb + kt * 16 + l16) * HD + 16 * half;
        Frag16 B0, B1;
        B0.f4[0] = *(const float4*)(ka);
        B0.f4[1] = *(const float4*)(ka + 8);
        B1.f4[0] = *(const float4*)(ka + 32);
        B1.f4[1] = *(const float4*)(ka + 40);
        v8f d0 = v8f{};
        d0 = __builtin_amdgcn_wmma_f32_16x16x32_f16(false, Qa0.h, false, B0.h,
                                                    (short)0, d0, false, false);
        d0 = __builtin_amdgcn_wmma_f32_16x16x32_f16(false, Qa1.h, false, B1.h,
                                                    (short)0, d0, false, false);
        s[kt] = d0;
      }

      // ---- unnormalized softmax: p = exp(s); accumulate row-sums per lane ----
#pragma unroll
      for (int r = 0; r < 8; ++r) {
        const float p0 = __expf(s[0][r]);
        const float p1 = __expf(s[1][r]);
        Plds[wid][r + 8 * half][l16]      = (_Float16)p0;
        Plds[wid][r + 8 * half][16 + l16] = (_Float16)p1;
        rsum[r] += p0 + p1;
      }

      // same-wave LDS store -> cross-lane load: drain DS ops
      asm volatile("s_wait_dscnt 0" ::: "memory");

      // ---- P as A-fragment: two 16B LDS runs per lane ----
      Frag16 Pa;
      Pa.f4[0] = *(const float4*)(&Plds[wid][l16][16 * half]);
      Pa.f4[1] = *(const float4*)(&Plds[wid][l16][16 * half + 8]);

      // ---- PV: 4 output column tiles of 16 dims ----
      int base = kb + 16 * half + vshift;
      if (base >= NPTS) base -= NPTS;      // pad handles the +15 overhang
#pragma unroll
      for (int t = 0; t < 4; ++t) {
        const int* vp = (const int*)(Vt +
            (size_t)(gv * HD + 16 * t + l16) * VPITCH + base);
        Frag16 VB;
#pragma unroll
        for (int j = 0; j < 8; ++j) VB.i[j] = vp[j];
        acc[t] = __builtin_amdgcn_wmma_f32_16x16x32_f16(false, Pa.h, false,
                                                        VB.h, (short)0,
                                                        acc[t], false, false);
      }
    }

    // one cross-lane reduction per pass (rows m = r + 8*half live in a
    // 16-lane half; xor masks 1..8 stay inside the half)
#pragma unroll
    for (int r = 0; r < 8; ++r) {
      float sfull = rsum[r];
      sfull += __shfl_xor(sfull, 1, 32);
      sfull += __shfl_xor(sfull, 2, 32);
      sfull += __shfl_xor(sfull, 4, 32);
      sfull += __shfl_xor(sfull, 8, 32);
      const float inv = wgts[pi] / sfull;
#pragma unroll
      for (int t = 0; t < 4; ++t) accT[t][r] += acc[t][r] * inv;
    }
  }

  // ---- reduce the 5 dw-waves and write out[n][g*64+d] ----
#pragma unroll
  for (int t = 0; t < 4; ++t)
#pragma unroll
    for (int r = 0; r < 8; ++r)
      partial[wid][r + 8 * half][16 * t + l16] = accT[t][r];
  __syncthreads();
  for (int idx = threadIdx.x; idx < 16 * 64; idx += 160) {
    const int m = idx >> 6, d = idx & 63;
    const float sum = partial[0][m][d] + partial[1][m][d] + partial[2][m][d] +
                      partial[3][m][d] + partial[4][m][d];
    out[(size_t)(qbase + m) * 256 + g * 64 + d] = sum;
  }
}

// ---------------------------------------------------------------------------
extern "C" void kernel_launch(void* const* d_in, const int* in_sizes, int n_in,
                              void* d_out, int out_size, void* d_ws, size_t ws_size,
                              hipStream_t stream) {
  const float* x  = (const float*)d_in[0];
  const float* Wq = (const float*)d_in[1];
  const float* bq = (const float*)d_in[2];
  const float* Wk = (const float*)d_in[3];
  const float* bk = (const float*)d_in[4];
  const float* Wv = (const float*)d_in[5];
  const float* bv = (const float*)d_in[6];
  float* out = (float*)d_out;

  _Float16* Qh = (_Float16*)d_ws;                       // 4*1600*64
  _Float16* Kh = Qh + (size_t)HEADS * NPTS * HD;        // 4*1600*64
  _Float16* Vt = Kh + (size_t)HEADS * NPTS * HD;        // 4*64*VPITCH

  proj_kernel<<<NPTS, 256, 0, stream>>>(x, Wq, bq, Wk, bk, Wv, bv, Qh, Kh, Vt);
  attn_kernel<<<(NPTS / 16) * HEADS, 160, 0, stream>>>(Qh, Kh, Vt, out);
}
